// MultiHeadSelfAttention_1279900254277
// MI455X (gfx1250) — matrix operand order repaired and checked by reading, not compiled, not run on hardware
//
#include <hip/hip_runtime.h>

typedef __attribute__((ext_vector_type(4)))  _Float16 v4h;
typedef __attribute__((ext_vector_type(8)))  _Float16 v8h;
typedef __attribute__((ext_vector_type(16))) _Float16 v16h;
typedef __attribute__((ext_vector_type(4)))  float    v4f;
typedef __attribute__((ext_vector_type(8)))  float    v8f;
typedef __attribute__((ext_vector_type(4)))  int      v4i;

#if defined(__HIP_DEVICE_COMPILE__) && __has_builtin(__builtin_amdgcn_global_load_async_to_lds_b128)
#define USE_ASYNC_LDS 1
typedef __attribute__((address_space(1))) v4i as1_v4i;  // global int4
typedef __attribute__((address_space(3))) v4i as3_v4i;  // LDS int4
#endif

// Build a 16-half A/B fragment from two 16-byte-aligned runs of 8 halfs.
__device__ __forceinline__ v16h ld16(const _Float16* lo, const _Float16* hi8) {
    v8h l = *(const v8h*)lo;
    v8h h = *(const v8h*)hi8;
    v16h r;
#pragma unroll
    for (int i = 0; i < 8; ++i) { r[i] = l[i]; r[i + 8] = h[i]; }
    return r;
}

__device__ __forceinline__ v8f wmma16(v16h a, v16h b, v8f c) {
    return __builtin_amdgcn_wmma_f32_16x16x32_f16(
        /*neg_a=*/false, a, /*neg_b=*/false, b,
        /*c_mod=*/(short)0, c, /*reuse_a=*/false, /*reuse_b=*/false);
}

// ---------------------------------------------------------------------------
// One-time f32 -> f16 conversion (vectorized 4-wide).
// ---------------------------------------------------------------------------
__global__ __launch_bounds__(256) void f32_to_f16(
    const v4f* __restrict__ src, v4h* __restrict__ dst, int n4)
{
    int i = blockIdx.x * 256 + threadIdx.x;
    if (i < n4) {
        v4f v = src[i];
        v4h h;
#pragma unroll
        for (int j = 0; j < 4; ++j) h[j] = (_Float16)v[j];
        dst[i] = h;
    }
}

// ---------------------------------------------------------------------------
// GEMM: out[M,N] = A[M,K] @ B[K,N] + bias.  A, B f16; accumulate f32 WMMA.
// Block = 256 threads = 8 waves; block tile 128(M) x 128(N); K step 32.
// Wave w computes rows [16w,16w+16), all 128 columns (8 accumulators).
// ---------------------------------------------------------------------------
__global__ __launch_bounds__(256) void gemm_f16wmma(
    const _Float16* __restrict__ A, const _Float16* __restrict__ B,
    const float* __restrict__ bias,
    _Float16* __restrict__ outH, float* __restrict__ outF,
    int M, int N, int K)
{
    __shared__ __attribute__((aligned(16))) _Float16 Asm_[128][40]; // [m][k], 80B pitch
    __shared__ __attribute__((aligned(16))) _Float16 Bsm [128][40]; // [n][k] transposed

    const int m0 = blockIdx.x * 128;
    const int n0 = blockIdx.y * 128;
    const int tid  = threadIdx.x;
    const int lane = tid & 31;
    const int wv   = tid >> 5;
    const int nn   = lane & 15;
    const int hi   = lane >> 4;

    v8f acc[8] = {};

    for (int k0 = 0; k0 < K; k0 += 32) {
        // Prefetch next K-tile (global_prefetch_b8).
        if (k0 + 32 < K) {
            __builtin_prefetch(A + (size_t)(m0 + (tid >> 1)) * K + (k0 + 32) + (tid & 1) * 16, 0, 3);
            __builtin_prefetch(B + (size_t)(k0 + 32 + (tid >> 3)) * N + n0 + (tid & 7) * 16, 0, 3);
        }
        // Stage A tile 128x32 row-major: 512 16B chunks / 256 threads.
#pragma unroll
        for (int i = tid; i < 512; i += 256) {
            int m = i >> 2, kc = (i & 3) * 8;
            *(v8h*)&Asm_[m][kc] = *(const v8h*)(A + (size_t)(m0 + m) * K + k0 + kc);
        }
        // Stage B tile 32x128 transposed -> Bsm[n][k]; vector read, scalar stores.
#pragma unroll
        for (int i = tid; i < 512; i += 256) {
            int kk = i >> 4, nc = (i & 15) * 8;
            v8h v = *(const v8h*)(B + (size_t)(k0 + kk) * N + n0 + nc);
#pragma unroll
            for (int j = 0; j < 8; ++j) Bsm[nc + j][kk] = v[j];
        }
        __syncthreads();

        // A frag: a[e] = A[m][ (e&7) | hi<<3 | (e>>3)<<4 ]
        v16h a = ld16(&Asm_[16 * wv + nn][8 * hi], &Asm_[16 * wv + nn][16 + 8 * hi]);
#pragma unroll
        for (int c = 0; c < 8; ++c) {
            v16h b = ld16(&Bsm[16 * c + nn][8 * hi], &Bsm[16 * c + nn][16 + 8 * hi]);
            acc[c] = wmma16(a, b, acc[c]);
        }
        __syncthreads();
    }

    // C layout: lane holds rows r+8*hi, column nn (+16c+n0); add bias, store.
#pragma unroll
    for (int c = 0; c < 8; ++c) {
        int col = n0 + 16 * c + nn;
        float bb = bias[col];
#pragma unroll
        for (int r = 0; r < 8; ++r) {
            int row = m0 + 16 * wv + r + 8 * hi;
            float v = acc[c][r] + bb;
            if (outH) outH[(size_t)row * N + col] = (_Float16)v;
            if (outF) outF[(size_t)row * N + col] = v;
        }
    }
}

// ---------------------------------------------------------------------------
// Flash attention: one (b, h, 64 q-rows) per 128-thread block (4 waves).
// K staged via async global->LDS (ASYNCcnt) when available; V transposed
// manually. Online softmax; P bounced via LDS (C-layout -> A-layout).
// ---------------------------------------------------------------------------
__global__ __launch_bounds__(128) void attn_flash(
    const _Float16* __restrict__ Qh, const _Float16* __restrict__ Kh,
    const _Float16* __restrict__ Vh, _Float16* __restrict__ Ch)
{
    const int S = 2048, E = 1024;
    const int bh = blockIdx.y;
    const int b  = bh >> 4;
    const int h  = bh & 15;
    const int q0 = blockIdx.x * 64;

    const int lane = threadIdx.x & 31;
    const int wv   = threadIdx.x >> 5;
    const int nn   = lane & 15;
    const int hi   = lane >> 4;

    __shared__ __attribute__((aligned(16))) _Float16 Ksm[32][72];      // [key][d], 144B pitch
    __shared__ __attribute__((aligned(16))) _Float16 Vts[64][40];      // [d][key], 80B pitch
    __shared__ __attribute__((aligned(16))) _Float16 Psm[4][16][40];   // per-wave probs

    const _Float16* qb = Qh + ((size_t)(b * S + q0 + 16 * wv + nn)) * E + h * 64;
    v16h aq0 = ld16(qb + 8 * hi,      qb + 16 + 8 * hi);   // d = 0..31
    v16h aq1 = ld16(qb + 32 + 8 * hi, qb + 48 + 8 * hi);   // d = 32..63

    const _Float16* kb = Kh + ((size_t)(b * S)) * E + h * 64;
    const _Float16* vb = Vh + ((size_t)(b * S)) * E + h * 64;

    v8f o[4] = {};
    float rm[8], rl[8];
#pragma unroll
    for (int r = 0; r < 8; ++r) { rm[r] = -1e30f; rl[r] = 0.f; }

    const float scale = 0.125f; // 1/sqrt(64)

    for (int k0 = 0; k0 < S; k0 += 32) {
#if defined(USE_ASYNC_LDS)
        // K tile 32x64 halfs = 256 x 16B chunks; per-lane async copies to LDS.
#pragma unroll
        for (int i = threadIdx.x; i < 256; i += 128) {
            int kk = i >> 3, dc = (i & 7) * 8;
            __builtin_amdgcn_global_load_async_to_lds_b128(
                (as1_v4i*)(kb + (size_t)(k0 + kk) * E + dc),
                (as3_v4i*)&Ksm[kk][dc], 0, 0);
        }
#else
#pragma unroll
        for (int i = threadIdx.x; i < 32 * 64; i += 128) {
            int kk = i >> 6, d = i & 63;
            Ksm[kk][d] = kb[(size_t)(k0 + kk) * E + d];
        }
#endif
        // V tile transposed (manual: transpose can't be expressed as 16B copies).
#pragma unroll
        for (int i = threadIdx.x; i < 32 * 64; i += 128) {
            int kk = i >> 6, d = i & 63;
            Vts[d][kk] = vb[(size_t)(k0 + kk) * E + d];
        }
#if defined(USE_ASYNC_LDS)
#if __has_builtin(__builtin_amdgcn_s_wait_asynccnt)
        __builtin_amdgcn_s_wait_asynccnt(0);
#else
        asm volatile("s_wait_asynccnt 0" ::: "memory");
#endif
#endif
        __syncthreads();

        // Scores S[16q x 32k]: two 16x16 tiles, d=64 -> 2 chained WMMAs each.
        v8f s0 = {}, s1 = {};
        {
            v16h b0 = ld16(&Ksm[nn][8 * hi],      &Ksm[nn][16 + 8 * hi]);
            s0 = wmma16(aq0, b0, s0);
            v16h b1 = ld16(&Ksm[nn][32 + 8 * hi], &Ksm[nn][48 + 8 * hi]);
            s0 = wmma16(aq1, b1, s0);
            v16h b2 = ld16(&Ksm[16 + nn][8 * hi],      &Ksm[16 + nn][16 + 8 * hi]);
            s1 = wmma16(aq0, b2, s1);
            v16h b3 = ld16(&Ksm[16 + nn][32 + 8 * hi], &Ksm[16 + nn][48 + 8 * hi]);
            s1 = wmma16(aq1, b3, s1);
        }

        // Online softmax; lane holds rows r+8*hi, cols nn and nn+16.
#pragma unroll
        for (int r = 0; r < 8; ++r) {
            float x0 = s0[r] * scale;
            float x1 = s1[r] * scale;
            float mx = fmaxf(x0, x1);
#pragma unroll
            for (int msk = 1; msk < 16; msk <<= 1)
                mx = fmaxf(mx, __shfl_xor(mx, msk, 32));
            float mnew  = fmaxf(rm[r], mx);
            float alpha = __expf(rm[r] - mnew);
            float p0 = __expf(x0 - mnew);
            float p1 = __expf(x1 - mnew);
            float rs = p0 + p1;
#pragma unroll
            for (int msk = 1; msk < 16; msk <<= 1)
                rs += __shfl_xor(rs, msk, 32);
            rl[r] = rl[r] * alpha + rs;
            rm[r] = mnew;
#pragma unroll
            for (int c = 0; c < 4; ++c) o[c][r] *= alpha;
            Psm[wv][r + 8 * hi][nn]      = (_Float16)p0;
            Psm[wv][r + 8 * hi][nn + 16] = (_Float16)p1;
        }
        __syncthreads(); // P visible before A-layout reads

        // P(16x32) @ V(32x64).
        v16h pa = ld16(&Psm[wv][nn][8 * hi], &Psm[wv][nn][16 + 8 * hi]);
#pragma unroll
        for (int c = 0; c < 4; ++c) {
            v16h vbq = ld16(&Vts[16 * c + nn][8 * hi], &Vts[16 * c + nn][16 + 8 * hi]);
            o[c] = wmma16(pa, vbq, o[c]);
        }
        __syncthreads(); // protect Ksm/Vts before next tile's fill
    }

    // Normalize and write concat [B, S, H*D] (f16).
    float inv[8];
#pragma unroll
    for (int r = 0; r < 8; ++r) inv[r] = 1.0f / rl[r];
#pragma unroll
    for (int c = 0; c < 4; ++c) {
#pragma unroll
        for (int r = 0; r < 8; ++r) {
            int row = q0 + 16 * wv + r + 8 * hi;
            Ch[((size_t)(b * S + row)) * E + h * 64 + 16 * c + nn] =
                (_Float16)(o[c][r] * inv[r]);
        }
    }
}

// ---------------------------------------------------------------------------
extern "C" void kernel_launch(void* const* d_in, const int* in_sizes, int n_in,
                              void* d_out, int out_size, void* d_ws, size_t ws_size,
                              hipStream_t stream) {
    const float* x  = (const float*)d_in[0];
    const float* Wq = (const float*)d_in[1];
    const float* bq = (const float*)d_in[2];
    const float* Wk = (const float*)d_in[3];
    const float* bk = (const float*)d_in[4];
    const float* Wv = (const float*)d_in[5];
    const float* bv = (const float*)d_in[6];
    const float* Wo = (const float*)d_in[7];
    const float* bo = (const float*)d_in[8];
    float* out = (float*)d_out;

    const int B = 2, S = 2048, E = 1024;
    const int M = B * S;                 // 4096 token rows
    const size_t ME = (size_t)M * E;     // 4M
    const size_t EE = (size_t)E * E;     // 1M

    _Float16* xh  = (_Float16*)d_ws;
    _Float16* Wqh = xh  + ME;
    _Float16* Wkh = Wqh + EE;
    _Float16* Wvh = Wkh + EE;
    _Float16* Woh = Wvh + EE;
    _Float16* Qh  = Woh + EE;
    _Float16* Kh  = Qh  + ME;
    _Float16* Vh  = Kh  + ME;
    _Float16* Ch  = Vh  + ME;

    // One-time f32 -> f16 conversions.
    {
        int n4 = (int)(ME / 4);
        f32_to_f16<<<(n4 + 255) / 256, 256, 0, stream>>>((const v4f*)x, (v4h*)xh, n4);
        int w4 = (int)(EE / 4);
        f32_to_f16<<<(w4 + 255) / 256, 256, 0, stream>>>((const v4f*)Wq, (v4h*)Wqh, w4);
        f32_to_f16<<<(w4 + 255) / 256, 256, 0, stream>>>((const v4f*)Wk, (v4h*)Wkh, w4);
        f32_to_f16<<<(w4 + 255) / 256, 256, 0, stream>>>((const v4f*)Wv, (v4h*)Wvh, w4);
        f32_to_f16<<<(w4 + 255) / 256, 256, 0, stream>>>((const v4f*)Wo, (v4h*)Woh, w4);
    }

    dim3 gGemm(M / 128, E / 128); // 32 x 8 blocks
    dim3 bGemm(256);

    // Q/K/V projections (f16 in, f16 out, bias fused).
    gemm_f16wmma<<<gGemm, bGemm, 0, stream>>>(xh, Wqh, bq, Qh, (float*)nullptr, M, E, E);
    gemm_f16wmma<<<gGemm, bGemm, 0, stream>>>(xh, Wkh, bk, Kh, (float*)nullptr, M, E, E);
    gemm_f16wmma<<<gGemm, bGemm, 0, stream>>>(xh, Wvh, bv, Vh, (float*)nullptr, M, E, E);

    // Flash attention: grid = (q-blocks of 64, B*H).
    attn_flash<<<dim3(S / 64, B * 16), dim3(128), 0, stream>>>(Qh, Kh, Vh, Ch);

    // Output projection (f16 concat in, f32 out to d_out).
    gemm_f16wmma<<<gGemm, bGemm, 0, stream>>>(Ch, Woh, bo, (_Float16*)nullptr, out, M, E, E);
}
